// FlowGuidedTSA_16844861735212
// MI455X (gfx1250) — matrix operand order repaired and checked by reading, not compiled, not run on hardware
//
#include <hip/hip_runtime.h>
#include <hip/hip_bf16.h>

// ---------------- problem constants (fixed by setup_inputs) ----------------
#define BS      8
#define NQ      1200
#define EMBED   256
#define HEADS   8
#define DIM     32
#define LEVELS  4
#define POINTS  4
#define NV      19560   // 92*160 + 46*80 + 23*40 + 12*20
#define NT      4       // N-tiles (16 cols each) per wave -> 16x64 strip

__constant__ const int LVL_H[LEVELS]     = {92, 46, 23, 12};
__constant__ const int LVL_W[LEVELS]     = {160, 80, 40, 20};
__constant__ const int LVL_START[LEVELS] = {0, 14720, 18400, 19320};

typedef __attribute__((ext_vector_type(16))) __bf16 v16bf;
typedef __attribute__((ext_vector_type(8)))  float  v8f;

// ---------------------------------------------------------------------------
// q = query + query_pos
// ---------------------------------------------------------------------------
__global__ void qadd_kernel(const float* __restrict__ a, const float* __restrict__ b,
                            float* __restrict__ out, int n) {
    int i = blockIdx.x * blockDim.x + threadIdx.x;
    if (i < n) out[i] = a[i] + b[i];
}

// ---------------------------------------------------------------------------
// Pre-swizzle an f32 weight matrix B[K,N] into bf16 WMMA B-fragment layout:
// tile (kt, tn) -> 512 contiguous bf16: lane L owns 16 elems (e = 2j+half):
//   K = kt*32 + 16*(L/16) + 2j + half, N = tn*16 + (L&15)
// so the GEMM inner loop loads one contiguous 32B chunk per lane per tile.
// ---------------------------------------------------------------------------
__global__ void bswizzle_kernel(const float* __restrict__ B, __bf16* __restrict__ Bsw,
                                int K, int N) {
    const int wid  = (blockIdx.x * blockDim.x + threadIdx.x) >> 5;
    const int lane = threadIdx.x & 31;
    const int tilesN = N >> 4;
    const int total  = (K >> 5) * tilesN;
    if (wid >= total) return;
    const int kt = wid / tilesN;
    const int tn = wid - kt * tilesN;
    const int kg = lane >> 4;
    const int n  = tn * 16 + (lane & 15);
    __bf16* dst = Bsw + ((size_t)wid << 9) + lane * 16;
    #pragma unroll
    for (int e = 0; e < 16; ++e) {
        int j = e >> 1, half = e & 1;
        int kk = (kt << 5) + (kg << 4) + (j << 1) + half;
        dst[e] = (__bf16)B[(size_t)kk * N + n];
    }
}

// ---------------------------------------------------------------------------
// WMMA bf16 GEMM, one wave per 16x64 output strip (4 tiles, 4 WMMAs / K-step).
//   A[M,K] f32 row-major, async-staged global->LDS (double buffered, ASYNCcnt)
//   Bsw:   pre-swizzled bf16 fragments (see bswizzle_kernel)
//   mode 0: Cf[m*N+n] = acc + bias
//   mode 1: bf16 V permuted to [b, h, pixel, 32]      (M = BS*NV, N = 256)
//   mode 2: Cf[m*N+n] = acc + bias + resid[m*N+n]     (final proj + identity)
// ---------------------------------------------------------------------------
__global__ void __launch_bounds__(256)
wmma_gemm_kernel(const float* __restrict__ A,
                 const __bf16* __restrict__ Bsw,
                 const float* __restrict__ bias,
                 const float* __restrict__ resid,
                 float* __restrict__ Cf,
                 __bf16* __restrict__ Cbf,
                 int M, int N, int K, int mode) {
    // per-wave double-buffered A stage: 2 x (16 rows x 32 f32) = 4KB, 8 waves
    __shared__ char stage[8 * 2 * 2048];

    const int waveInBlk = threadIdx.x >> 5;
    const int lane      = threadIdx.x & 31;
    const int wid       = (blockIdx.x * blockDim.x + threadIdx.x) >> 5;
    const int stripsN   = N >> 6;                 // 64-wide strips
    const int totalStrips = (M >> 4) * stripsN;
    if (wid >= totalStrips) return;               // wave-uniform: EXEC stays all-1s

    const int tm  = wid / stripsN;
    const int tsn = wid - tm * stripsN;
    const int tilesN = N >> 4;
    const int tn0 = tsn << 2;
    const int kg  = lane >> 4;

    // LDS byte offset of this wave's stage (flat LDS addr keeps offset in [31:0])
    const unsigned lbase = (unsigned)(size_t)(&stage[0]) + (unsigned)(waveInBlk * 4096);
    const float* aRowBase = A + (size_t)(tm * 16) * K;

    // issue async copy of A tile [16 x 32 f32] for K-step base k0 into buffer
    auto issueA = [&](int k0, int bufsel) {
        #pragma unroll
        for (int it = 0; it < 4; ++it) {
            int t = it * 32 + lane;               // 128 x 16B transfers
            int r = t >> 3;                       // row 0..15
            int c = t & 7;                        // 16B chunk 0..7
            unsigned long long g =
                (unsigned long long)(aRowBase + (size_t)r * K + k0) + (unsigned)(c * 16);
            unsigned l = lbase + (unsigned)(bufsel * 2048 + r * 128 + c * 16);
            asm volatile("global_load_async_to_lds_b128 %0, %1, off"
                         :: "v"(l), "v"(g) : "memory");
        }
    };

    v8f acc[NT];
    #pragma unroll
    for (int t = 0; t < NT; ++t) acc[t] = (v8f){0.f,0.f,0.f,0.f,0.f,0.f,0.f,0.f};

    const int KSTEPS = K >> 5;
    issueA(0, 0);
    int cur = 0;
    for (int ks = 0; ks < KSTEPS; ++ks) {
        if (ks + 1 < KSTEPS) {
            issueA((ks + 1) << 5, cur ^ 1);       // prefetch next tile
            asm volatile("s_wait_asynccnt 0x4" ::: "memory");  // current tile done
        } else {
            asm volatile("s_wait_asynccnt 0x0" ::: "memory");
        }

        // A 16x32 bf16 fragment from LDS (contiguous k = 16*(lane/16) + e, matching Bsw)
        const float* aTile = (const float*)(&stage[0] + waveInBlk * 4096 + cur * 2048);
        const float* aRow  = aTile + (lane & 15) * 32;
        v16bf afrag;
        #pragma unroll
        for (int j = 0; j < 8; ++j) {
            int kk = (kg << 4) + (j << 1);
            float2 f = *(const float2*)(aRow + kk);
            afrag[2 * j]     = (__bf16)f.x;
            afrag[2 * j + 1] = (__bf16)f.y;
        }

        // 4 pre-swizzled B fragments, 1 contiguous 32B load each
        const __bf16* bK = Bsw + ((size_t)(ks * tilesN + tn0) << 9) + lane * 16;
        #pragma unroll
        for (int t = 0; t < NT; ++t) {
            v16bf bfrag = *(const v16bf*)(bK + ((size_t)t << 9));
            acc[t] = __builtin_amdgcn_wmma_f32_16x16x32_bf16(
                false, afrag, false, bfrag, (short)0, acc[t], false, false);
        }
        cur ^= 1;
    }

    // C/D layout: VGPR j -> M = 8*(lane/16) + j, N = lane&15
    #pragma unroll
    for (int t = 0; t < NT; ++t) {
        const int ncol = (tn0 + t) * 16 + (lane & 15);
        const float bn = bias ? bias[ncol] : 0.0f;
        #pragma unroll
        for (int j = 0; j < 8; ++j) {
            int mm  = tm * 16 + (kg << 3) + j;
            float v = acc[t][j] + bn;
            if (mode == 0) {
                Cf[(size_t)mm * N + ncol] = v;
            } else if (mode == 2) {
                Cf[(size_t)mm * N + ncol] = v + resid[(size_t)mm * N + ncol];
            } else {
                int bidx = mm / NV;
                int pix  = mm - bidx * NV;
                int h    = ncol >> 5;
                int d    = ncol & 31;
                Cbf[(((size_t)(bidx * HEADS + h) * NV + pix) << 5) + d] = (__bf16)v;
            }
        }
    }
}

// ---------------------------------------------------------------------------
// Softmax over 16 (level,point) logits per (b,q,h); in place on [BS*NQ, 128]
// ---------------------------------------------------------------------------
__global__ void softmax16_kernel(float* __restrict__ attn, int total) {
    int t = blockIdx.x * blockDim.x + threadIdx.x;
    if (t >= total) return;
    float* p = attn + (size_t)t * 16;
    float m = p[0];
    #pragma unroll
    for (int i = 1; i < 16; ++i) m = fmaxf(m, p[i]);
    float s = 0.0f;
    float e[16];
    #pragma unroll
    for (int i = 0; i < 16; ++i) { e[i] = __expf(p[i] - m); s += e[i]; }
    float inv = 1.0f / s;
    #pragma unroll
    for (int i = 0; i < 16; ++i) p[i] = e[i] * inv;
}

// ---------------------------------------------------------------------------
// Deformable sampler: one wave per (b, q, h); lane = channel d (DIM==wave32).
// Wave-uniform addresses/weights; coalesced 64B bf16 corner loads.
// grid_sample bilinear, zeros padding, align_corners=False.
// ---------------------------------------------------------------------------
__global__ void sampler_kernel(const __bf16* __restrict__ vperm,
                               const float* __restrict__ off,
                               const float* __restrict__ attn,
                               const float* __restrict__ ref,
                               const float* __restrict__ lastoff,
                               float* __restrict__ out_heads) {
    const int gid  = blockIdx.x * blockDim.x + threadIdx.x;
    const int w    = gid >> 5;
    const int d    = threadIdx.x & 31;
    const int total = BS * NQ * HEADS;
    if (w >= total) return;

    const int b   = w / (NQ * HEADS);
    const int rem = w - b * (NQ * HEADS);
    const int q   = rem / HEADS;
    const int h   = rem - q * HEADS;

    const size_t row    = (size_t)b * NQ + q;
    const float  lox    = lastoff[(row * HEADS + h) * 2 + 0];
    const float  loy    = lastoff[(row * HEADS + h) * 2 + 1];
    const float* offp   = off  + row * (HEADS * LEVELS * POINTS * 2) + h * (LEVELS * POINTS * 2);
    const float* attnp  = attn + row * (HEADS * LEVELS * POINTS)     + h * (LEVELS * POINTS);
    const __bf16* vbase = vperm + (((size_t)(b * HEADS + h)) * NV << 5);

    float acc = 0.0f;
    #pragma unroll
    for (int lvl = 0; lvl < LEVELS; ++lvl) {
        const int   W  = LVL_W[lvl];
        const int   H  = LVL_H[lvl];
        const int   st = LVL_START[lvl];
        const float rx = ref[(row * LEVELS + lvl) * 2 + 0];
        const float ry = ref[(row * LEVELS + lvl) * 2 + 1];
        const float fW = (float)W, fH = (float)H;

        #pragma unroll
        for (int p = 0; p < POINTS; ++p) {
            const float ox = offp[lvl * (POINTS * 2) + p * 2 + 0];
            const float oy = offp[lvl * (POINTS * 2) + p * 2 + 1];
            const float a  = attnp[lvl * POINTS + p];

            // loc = ref + off/(W,H) + last_off ; px = loc*W - 0.5
            const float px = (rx + ox / fW + lox) * fW - 0.5f;
            const float py = (ry + oy / fH + loy) * fH - 0.5f;

            const float x0f = floorf(px), y0f = floorf(py);
            const int   x0 = (int)x0f, y0 = (int)y0f, x1 = x0 + 1, y1 = y0 + 1;
            const float wx1 = px - x0f, wx0 = 1.0f - wx1;
            const float wy1 = py - y0f, wy0 = 1.0f - wy1;

            const bool vx0 = (x0 >= 0) & (x0 < W);
            const bool vx1 = (x1 >= 0) & (x1 < W);
            const bool vy0 = (y0 >= 0) & (y0 < H);
            const bool vy1 = (y1 >= 0) & (y1 < H);
            const int  cx0 = min(max(x0, 0), W - 1);
            const int  cx1 = min(max(x1, 0), W - 1);
            const int  cy0 = min(max(y0, 0), H - 1);
            const int  cy1 = min(max(y1, 0), H - 1);

            const float c00 = (float)vbase[(((size_t)(st + cy0 * W + cx0)) << 5) + d];
            const float c10 = (float)vbase[(((size_t)(st + cy0 * W + cx1)) << 5) + d];
            const float c01 = (float)vbase[(((size_t)(st + cy1 * W + cx0)) << 5) + d];
            const float c11 = (float)vbase[(((size_t)(st + cy1 * W + cx1)) << 5) + d];

            float s = c00 * (wx0 * wy0 * (vx0 && vy0 ? 1.0f : 0.0f))
                    + c10 * (wx1 * wy0 * (vx1 && vy0 ? 1.0f : 0.0f))
                    + c01 * (wx0 * wy1 * (vx0 && vy1 ? 1.0f : 0.0f))
                    + c11 * (wx1 * wy1 * (vx1 && vy1 ? 1.0f : 0.0f));
            acc = fmaf(a, s, acc);
        }
    }
    out_heads[row * EMBED + h * DIM + d] = acc;
}

// ---------------------------------------------------------------------------
extern "C" void kernel_launch(void* const* d_in, const int* in_sizes, int n_in,
                              void* d_out, int out_size, void* d_ws, size_t ws_size,
                              hipStream_t stream) {
    const float* query   = (const float*)d_in[0];
    const float* value   = (const float*)d_in[1];
    const float* qpos    = (const float*)d_in[2];
    const float* refpts  = (const float*)d_in[3];
    const float* lastoff = (const float*)d_in[4];
    // d_in[5] = spatial_shapes (fixed; hardcoded as constants)
    const float* Wv   = (const float*)d_in[6];
    const float* bv   = (const float*)d_in[7];
    const float* Wo   = (const float*)d_in[8];
    const float* bo   = (const float*)d_in[9];
    const float* Wa   = (const float*)d_in[10];
    const float* ba   = (const float*)d_in[11];
    const float* Wout = (const float*)d_in[12];
    const float* bout = (const float*)d_in[13];
    float* out = (float*)d_out;
    (void)ws_size; (void)in_sizes; (void)n_in; (void)out_size;

    // ---- workspace carve-up (256B-aligned sections) ----
    char* ws = (char*)d_ws;
    const size_t vperm_bytes = (size_t)BS * HEADS * NV * DIM * sizeof(__bf16); // 80,117,760
    const size_t q_bytes     = (size_t)BS * NQ * EMBED * sizeof(float);        //  9,830,400
    const size_t off_bytes   = (size_t)BS * NQ * 512 * sizeof(float);          // 19,660,800
    const size_t attn_bytes  = (size_t)BS * NQ * 128 * sizeof(float);          //  4,915,200
    const size_t oh_bytes    = (size_t)BS * NQ * EMBED * sizeof(float);        //  9,830,400
    const size_t bswV_bytes  = (size_t)(256/32) * (256/16) * 512 * 2;          //    131,072
    const size_t bswO_bytes  = (size_t)(256/32) * (512/16) * 512 * 2;          //    262,144
    const size_t bswA_bytes  = (size_t)(256/32) * (128/16) * 512 * 2;          //     65,536

    size_t o = 0;
    __bf16* vperm  = (__bf16*)(ws + o); o += vperm_bytes;
    float*  qbuf   = (float*) (ws + o); o += q_bytes;
    float*  offbuf = (float*) (ws + o); o += off_bytes;
    float*  attbuf = (float*) (ws + o); o += attn_bytes;
    float*  oheads = (float*) (ws + o); o += oh_bytes;
    __bf16* bswV   = (__bf16*)(ws + o); o += bswV_bytes;
    __bf16* bswO   = (__bf16*)(ws + o); o += bswO_bytes;
    __bf16* bswA   = (__bf16*)(ws + o); o += bswA_bytes;
    __bf16* bswOut = (__bf16*)(ws + o);

    const int BLK = 256;               // 8 wave32 per block
    const int WPB = BLK / 32;

    // 1) q = query + query_pos
    {
        int n = BS * NQ * EMBED;
        qadd_kernel<<<(n + BLK - 1) / BLK, BLK, 0, stream>>>(query, qpos, qbuf, n);
    }
    // 2) pre-swizzle all weight matrices into bf16 B-fragment layout
    {
        int tWv = (256/32)*(256/16), tWo = (256/32)*(512/16),
            tWa = (256/32)*(128/16), tWu = (256/32)*(256/16);
        bswizzle_kernel<<<(tWv + WPB - 1)/WPB, BLK, 0, stream>>>(Wv,   bswV,   256, 256);
        bswizzle_kernel<<<(tWo + WPB - 1)/WPB, BLK, 0, stream>>>(Wo,   bswO,   256, 512);
        bswizzle_kernel<<<(tWa + WPB - 1)/WPB, BLK, 0, stream>>>(Wa,   bswA,   256, 128);
        bswizzle_kernel<<<(tWu + WPB - 1)/WPB, BLK, 0, stream>>>(Wout, bswOut, 256, 256);
    }
    // 3) V projection: value @ Wv + bv -> bf16 [b,h,pix,32]
    {
        int M = BS * NV, N = EMBED, K = EMBED;
        int strips = (M / 16) * (N / 64);
        wmma_gemm_kernel<<<(strips + WPB - 1) / WPB, BLK, 0, stream>>>(
            value, bswV, bv, nullptr, nullptr, vperm, M, N, K, 1);
    }
    // 4) offsets: q @ Wo + bo
    {
        int M = BS * NQ, N = 512, K = EMBED;
        int strips = (M / 16) * (N / 64);
        wmma_gemm_kernel<<<(strips + WPB - 1) / WPB, BLK, 0, stream>>>(
            qbuf, bswO, bo, nullptr, offbuf, nullptr, M, N, K, 0);
    }
    // 5) attention logits: q @ Wa + ba
    {
        int M = BS * NQ, N = 128, K = EMBED;
        int strips = (M / 16) * (N / 64);
        wmma_gemm_kernel<<<(strips + WPB - 1) / WPB, BLK, 0, stream>>>(
            qbuf, bswA, ba, nullptr, attbuf, nullptr, M, N, K, 0);
    }
    // 6) softmax over 16 (level,point) weights per head
    {
        int total = BS * NQ * HEADS;
        softmax16_kernel<<<(total + BLK - 1) / BLK, BLK, 0, stream>>>(attbuf, total);
    }
    // 7) deformable bilinear sampling + weighted accumulate
    {
        int threads = BS * NQ * HEADS * 32;
        sampler_kernel<<<(threads + BLK - 1) / BLK, BLK, 0, stream>>>(
            vperm, offbuf, attbuf, refpts, lastoff, oheads);
    }
    // 8) out = out_heads @ Wout + bout + query (identity residual)
    {
        int M = BS * NQ, N = EMBED, K = EMBED;
        int strips = (M / 16) * (N / 64);
        wmma_gemm_kernel<<<(strips + WPB - 1) / WPB, BLK, 0, stream>>>(
            oheads, bswOut, bout, query, out, nullptr, M, N, K, 2);
    }
}
